// Attention_87625922773691
// MI455X (gfx1250) — matrix operand order repaired and checked by reading, compiled, not run on hardware
//
#include <hip/hip_runtime.h>
#include <hip/hip_bf16.h>
#include <math.h>

// Problem constants (from the reference)
constexpr int B_ = 8;
constexpr int S_ = 4096;
constexpr int D_ = 512;
constexpr int L_ = 8921;

constexpr int TL  = 32;          // labels per workgroup
constexpr int CH  = 64;          // s-chunk staged in LDS
constexpr int DSU = D_ + 8;      // padded row stride (halfs) for [rows][D] tiles
constexpr int CHS = CH + 8;      // padded row stride (halfs) for [rows][CH] tiles
constexpr int NLT = (L_ + TL - 1) / TL;   // 279 label tiles

static_assert(S_ % CH == 0, "");
static_assert(D_ % 32 == 0, "");

typedef __attribute__((ext_vector_type(16))) _Float16 v16h;
typedef __attribute__((ext_vector_type(8)))  _Float16 v8h;
typedef __attribute__((ext_vector_type(4)))  _Float16 v4h;
typedef __attribute__((ext_vector_type(8)))  float    v8f;

// ---- Fragment loaders (ISA 7.12.2 layouts, wave32) -------------------------
// A matrix 16x32 f16: lane m=lane&15 holds row m.
//   elems 0..7  -> K = k0 + h*8 + j          (h = lane>>4)
//   elems 8..15 -> K = k0 + 16 + h*8 + j
__device__ __forceinline__ v16h ldA(const _Float16* base, int stride, int lane, int k) {
  const int m = lane & 15;
  const int h = (lane >> 4) & 1;
  const _Float16* p = base + m * stride + k + h * 8;
  v8h lo = *(const v8h*)p;         // 16B aligned: stride*2 is a multiple of 16
  v8h hi = *(const v8h*)(p + 16);
  v16h r;
#pragma unroll
  for (int i = 0; i < 8; ++i) { r[i] = lo[i]; r[i + 8] = hi[i]; }
  return r;
}

// B matrix 32x16 f16: lane n=lane&15 holds column n; same K order as A (elems 0..7 -> k0+h*8+j, 8..15 -> k0+16+h*8+j).
// Source must be B^T row-major: base[n*stride + K].
__device__ __forceinline__ v16h ldB(const _Float16* base, int stride, int lane, int k) {
  const int n = lane & 15;
  const int h = (lane >> 4) & 1;
  const _Float16* p = base + n * stride + k + h * 8;
  v8h lo = *(const v8h*)p;
  v8h hi = *(const v8h*)(p + 16);
  v16h r;
#pragma unroll
  for (int i = 0; i < 8; ++i) { r[i] = lo[i]; r[i + 8] = hi[i]; }
  return r;
}

__device__ __forceinline__ v8f wmma_f16(v16h a, v16h b, v8f c) {
  return __builtin_amdgcn_wmma_f32_16x16x32_f16(false, a, false, b, (short)0, c, false, false);
}

__global__ __launch_bounds__(256)
void caml_attn_kernel(const float* __restrict__ x, const float* __restrict__ U,
                      float* __restrict__ out, float* __restrict__ alpha) {
  // LDS: ~111 KB total -> 2 workgroups per WGP (320 KB pool)
  __shared__ alignas(16) _Float16 sU[TL * DSU];     // U tile, f16        (32.5 KB)
  __shared__ alignas(16) _Float16 sX[D_ * CHS];     // x chunk (phase1: [CH][DSU],
                                                    //          phase2: [D][CHS])  (72 KB)
  __shared__ alignas(16) _Float16 sA[TL * CHS];     // alpha chunk, f16   (4.5 KB)
  __shared__ float sRm[8 * TL];                     // per-wave running max
  __shared__ float sRl[8 * TL];                     // per-wave running sumexp
  __shared__ float sM[TL];
  __shared__ float sLinv[TL];

  const int tid  = threadIdx.x;
  const int lane = tid & 31;
  const int wave = tid >> 5;
  const int lt   = blockIdx.x;
  const int b    = blockIdx.y;
  const int l0   = lt * TL;
  const bool fullTile = (l0 + TL <= L_);   // uniform across the block (278/279)

  const float* xb  = x     + (size_t)b * S_ * D_;
  float*       alb = alpha + (size_t)b * L_ * S_;
  float*       ob  = out   + (size_t)b * L_ * D_;

  // ---- Stage U tile (f32 -> f16), zero-fill rows past L --------------------
  if (fullTile) {
    for (int i = tid; i < TL * (D_ / 4); i += 256) {
      const int r  = i >> 7;          // D_/4 == 128
      const int c4 = i & 127;
      float4 v = ((const float4*)(U + (size_t)(l0 + r) * D_))[c4];
      v4h h = {(_Float16)v.x, (_Float16)v.y, (_Float16)v.z, (_Float16)v.w};
      *(v4h*)&sU[r * DSU + c4 * 4] = h;
    }
  } else {
    for (int i = tid; i < TL * (D_ / 4); i += 256) {
      const int r  = i >> 7;
      const int c4 = i & 127;
      const int l  = l0 + r;
      float4 v = (l < L_) ? ((const float4*)(U + (size_t)l * D_))[c4]
                          : make_float4(0.f, 0.f, 0.f, 0.f);
      v4h h = {(_Float16)v.x, (_Float16)v.y, (_Float16)v.z, (_Float16)v.w};
      *(v4h*)&sU[r * DSU + c4 * 4] = h;
    }
  }

  // ==== Phase 1: raw scores = U_tile @ x^T, online row max / sumexp =========
  const int mt = wave >> 2;   // 0..1 : which 16-label half
  const int nt = wave & 3;    // 0..3 : which 16-col slice of the chunk

  float m_[8], l_[8];
#pragma unroll
  for (int e = 0; e < 8; ++e) { m_[e] = -INFINITY; l_[e] = 0.f; }

  const int rbase = (lane >> 4) * 8;             // C/D: row = mt*16 + rbase + e
  const int rrow0 = l0 + mt * 16 + rbase;        // first global row of my C half

  for (int c = 0; c < S_ / CH; ++c) {
    __syncthreads();
    // stage x chunk [CH][D] as f16, row-major
    for (int i = tid; i < CH * (D_ / 4); i += 256) {
      const int r  = i >> 7;
      const int c4 = i & 127;
      float4 v = ((const float4*)(xb + (size_t)(c * CH + r) * D_))[c4];
      v4h h = {(_Float16)v.x, (_Float16)v.y, (_Float16)v.z, (_Float16)v.w};
      *(v4h*)&sX[r * DSU + c4 * 4] = h;
      if (c + 1 < S_ / CH)   // gfx1250 global_prefetch_b8 for next chunk
        __builtin_prefetch(xb + (size_t)((c + 1) * CH + r) * D_ + 4 * c4, 0, 1);
    }
    __syncthreads();

    v8f acc = {};
#pragma unroll 8
    for (int k = 0; k < D_; k += 32) {
      v16h a  = ldA(&sU[mt * 16 * DSU], DSU, lane, k);
      v16h bf = ldB(&sX[nt * 16 * DSU], DSU, lane, k);   // B^T rows = s-columns
      acc = wmma_f16(a, bf, acc);
    }

    // write raw scores; base + unroll-constant offsets (fold into inst offset)
    const int scol = c * CH + nt * 16 + (lane & 15);
    if (fullTile) {
      float* p = alb + (size_t)rrow0 * S_ + scol;
#pragma unroll
      for (int e = 0; e < 8; ++e) p[e * S_] = acc[e];
    } else {
      float* p = alb + (size_t)rrow0 * S_ + scol;
#pragma unroll
      for (int e = 0; e < 8; ++e)
        if (rrow0 + e < L_) p[e * S_] = acc[e];
    }

    // online softmax stats: reduce across the 16 lanes holding one row
#pragma unroll
    for (int e = 0; e < 8; ++e) {
      float t = acc[e];
      t = fmaxf(t, __shfl_xor(t, 1));
      t = fmaxf(t, __shfl_xor(t, 2));
      t = fmaxf(t, __shfl_xor(t, 4));
      t = fmaxf(t, __shfl_xor(t, 8));
      const float nm = fmaxf(m_[e], t);
      float p = __expf(acc[e] - nm);
      p += __shfl_xor(p, 1);
      p += __shfl_xor(p, 2);
      p += __shfl_xor(p, 4);
      p += __shfl_xor(p, 8);
      l_[e] = l_[e] * __expf(m_[e] - nm) + p;
      m_[e] = nm;
    }
  }

  // ---- Combine stats across waves ------------------------------------------
  if ((lane & 15) == 0) {
#pragma unroll
    for (int e = 0; e < 8; ++e) {
      const int r = mt * 16 + rbase + e;
      sRm[wave * TL + r] = m_[e];
      sRl[wave * TL + r] = l_[e];
    }
  }
  __threadfence();     // raw-score stores visible before phase-2 reads
  __syncthreads();
  if (tid < TL) {
    const int r = tid, w0 = (r >> 4) * 4;
    float M = -INFINITY;
    for (int w = w0; w < w0 + 4; ++w) M = fmaxf(M, sRm[w * TL + r]);
    float Ls = 0.f;
    for (int w = w0; w < w0 + 4; ++w) Ls += sRl[w * TL + r] * __expf(sRm[w * TL + r] - M);
    sM[r]    = M;
    sLinv[r] = 1.f / Ls;
  }
  __syncthreads();

  // ==== Phase 2: alpha normalize+store, out = alpha @ x ======================
  const int nq = wave & 3;     // 128-wide column quarter of D
  v8f oacc[8];
#pragma unroll
  for (int j = 0; j < 8; ++j) oacc[j] = (v8f){};

  for (int c = 0; c < S_ / CH; ++c) {
    __syncthreads();
    // stage x chunk transposed: sX as [D][CHS] f16. Each thread: a 4x4 block,
    // 4 coalesced float4 loads -> register transpose -> 4x 8B LDS stores.
    for (int i = tid; i < (CH / 4) * (D_ / 4); i += 256) {
      const int sb = i >> 7;          // which 4-row s group (CH/4 = 16)
      const int db = i & 127;         // which 4-col d group (D/4 = 128)
      const int s0 = sb * 4, d0 = db * 4;
      const float* gp = xb + (size_t)(c * CH + s0) * D_ + d0;
      float4 r0 = ((const float4*)(gp + 0 * D_))[0];
      float4 r1 = ((const float4*)(gp + 1 * D_))[0];
      float4 r2 = ((const float4*)(gp + 2 * D_))[0];
      float4 r3 = ((const float4*)(gp + 3 * D_))[0];
      v4h c0 = {(_Float16)r0.x, (_Float16)r1.x, (_Float16)r2.x, (_Float16)r3.x};
      v4h c1 = {(_Float16)r0.y, (_Float16)r1.y, (_Float16)r2.y, (_Float16)r3.y};
      v4h c2 = {(_Float16)r0.z, (_Float16)r1.z, (_Float16)r2.z, (_Float16)r3.z};
      v4h c3 = {(_Float16)r0.w, (_Float16)r1.w, (_Float16)r2.w, (_Float16)r3.w};
      *(v4h*)&sX[(d0 + 0) * CHS + s0] = c0;
      *(v4h*)&sX[(d0 + 1) * CHS + s0] = c1;
      *(v4h*)&sX[(d0 + 2) * CHS + s0] = c2;
      *(v4h*)&sX[(d0 + 3) * CHS + s0] = c3;
    }
    // normalize raw scores -> alpha (f32 out) and f16 tile in LDS
    if (fullTile) {
      for (int i = tid; i < TL * CH; i += 256) {
        const int r  = i >> 6;        // CH == 64
        const int cc = i & 63;
        const size_t idx = (size_t)(l0 + r) * S_ + c * CH + cc;
        const float a = __expf(alb[idx] - sM[r]) * sLinv[r];
        alb[idx] = a;
        sA[r * CHS + cc] = (_Float16)a;
      }
    } else {
      for (int i = tid; i < TL * CH; i += 256) {
        const int r  = i >> 6;
        const int cc = i & 63;
        const int l  = l0 + r;
        float a = 0.f;
        if (l < L_) {
          const size_t idx = (size_t)l * S_ + c * CH + cc;
          a = __expf(alb[idx] - sM[r]) * sLinv[r];
          alb[idx] = a;
        }
        sA[r * CHS + cc] = (_Float16)a;
      }
    }
    __syncthreads();

    v16h a0 = ldA(&sA[mt * 16 * CHS], CHS, lane, 0);
    v16h a1 = ldA(&sA[mt * 16 * CHS], CHS, lane, 32);
#pragma unroll
    for (int j = 0; j < 8; ++j) {
      const int dcol = nq * 128 + j * 16;
      v16h b0 = ldB(&sX[dcol * CHS], CHS, lane, 0);
      v16h b1 = ldB(&sX[dcol * CHS], CHS, lane, 32);
      oacc[j] = wmma_f16(a0, b0, oacc[j]);
      oacc[j] = wmma_f16(a1, b1, oacc[j]);
    }
  }

  // ---- store out tile: base + unroll-constant offsets ----------------------
  {
    float* p = ob + (size_t)rrow0 * D_ + nq * 128 + (lane & 15);
    if (fullTile) {
#pragma unroll
      for (int j = 0; j < 8; ++j)
#pragma unroll
        for (int e = 0; e < 8; ++e) p[e * D_ + j * 16] = oacc[j][e];
    } else {
#pragma unroll
      for (int e = 0; e < 8; ++e)
        if (rrow0 + e < L_) {
#pragma unroll
          for (int j = 0; j < 8; ++j) p[e * D_ + j * 16] = oacc[j][e];
        }
    }
  }
}

extern "C" void kernel_launch(void* const* d_in, const int* in_sizes, int n_in,
                              void* d_out, int out_size, void* d_ws, size_t ws_size,
                              hipStream_t stream) {
  (void)in_sizes; (void)n_in; (void)d_ws; (void)ws_size; (void)out_size;
  const float* x = (const float*)d_in[0];
  const float* U = (const float*)d_in[1];
  float* out   = (float*)d_out;
  float* alpha = out + (size_t)B_ * L_ * D_;   // tuple order: (out, alpha)
  dim3 grid(NLT, B_, 1);
  hipLaunchKernelGGL(caml_attn_kernel, grid, dim3(256, 1, 1), 0, stream, x, U, out, alpha);
}
